// video_header_15333033247313
// MI455X (gfx1250) — matrix operand order repaired and checked by reading, not compiled, not run on hardware
//
#include <hip/hip_runtime.h>
#include <hip/hip_bf16.h>
#include <math.h>

// ---------------------------------------------------------------------------
// CDNA5 (gfx1250) bf16 WMMA implementation of the MoE spatio-temporal block.
// - All matmuls via v_wmma_f32_16x16x32_bf16
// - Weights converted fp32->bf16 once per launch
// - GEMM main loop: double-buffered LDS fed by GLOBAL_LOAD_ASYNC_TO_LDS_B128
//   (ASYNCcnt pipeline), so TDM-style copies overlap WMMA issue.
// ---------------------------------------------------------------------------

typedef __attribute__((ext_vector_type(16))) __bf16 v16bf;
typedef __attribute__((ext_vector_type(8)))  __bf16 v8bf;
typedef __attribute__((ext_vector_type(8)))  float  v8f;

__device__ __forceinline__ float bf2f(__bf16 h) { return (float)h; }
__device__ __forceinline__ __bf16 f2bf(float f) { return (__bf16)f; }

__device__ __forceinline__ float gelu_f(float x) {
  return 0.5f * x * (1.0f + erff(x * 0.70710678118654752f));
}

// ---- async global->LDS (gfx1250): one wave-instruction moves 32x16B --------
#define USE_ASYNC_LDS 1
#if USE_ASYNC_LDS
__device__ __forceinline__ void async_ld_b128(void* lds, const void* gptr) {
  // low 32 bits of a generic LDS pointer are the LDS byte address
  unsigned loff = (unsigned)(size_t)lds;
  asm volatile("global_load_async_to_lds_b128 %0, %1, off"
               :: "v"(loff), "v"(gptr) : "memory");
}
#define WAIT_ASYNC(n) asm volatile("s_wait_asynccnt " #n ::: "memory")
#else
__device__ __forceinline__ void async_ld_b128(void* lds, const void* gptr) {
  *(v8bf*)lds = *(const v8bf*)gptr;
}
#define WAIT_ASYNC(n)
#endif

// ---- WMMA fragment loaders (contiguous k = 16*(lane/16) + s for A and B, wave32) --------------------
__device__ __forceinline__ v16bf load_frag_a(const __bf16* lds, int ldk,
                                             int row_base, int lane) {
  int m  = row_base + (lane & 15);
  int kb = (lane & 16) ? 16 : 0;
  const __bf16* p = lds + m * ldk + kb;
  v8bf lo = *(const v8bf*)(p);
  v8bf hi = *(const v8bf*)(p + 8);
  v16bf a;
#pragma unroll
  for (int i = 0; i < 8; ++i) { a[i] = lo[i]; a[i + 8] = hi[i]; }
  return a;
}

__device__ __forceinline__ v16bf load_frag_b(const __bf16* lds, int ldk,
                                             int col_base, int lane) {
  int n  = col_base + (lane & 15);
  int kb = (lane & 16) ? 16 : 0;
  const __bf16* p = lds + n * ldk + kb;
  v8bf lo = *(const v8bf*)(p);
  v8bf hi = *(const v8bf*)(p + 8);
  v16bf b;
#pragma unroll
  for (int i = 0; i < 8; ++i) { b[i] = lo[i]; b[i + 8] = hi[i]; }
  return b;
}

// ---------------------------------------------------------------------------
// One-shot fp32 -> bf16 conversion (weights)
// ---------------------------------------------------------------------------
__global__ void cvt_bf16_kernel(const float* __restrict__ src,
                                __bf16* __restrict__ dst, int n) {
  const int i = (blockIdx.x * blockDim.x + threadIdx.x) * 8;
  if (i >= n) return;
  const float4 a = *(const float4*)(src + i);
  const float4 b = *(const float4*)(src + i + 4);
  v8bf o;
  o[0] = f2bf(a.x); o[1] = f2bf(a.y); o[2] = f2bf(a.z); o[3] = f2bf(a.w);
  o[4] = f2bf(b.x); o[5] = f2bf(b.y); o[6] = f2bf(b.z); o[7] = f2bf(b.w);
  *(v8bf*)(dst + i) = o;
}

// ---------------------------------------------------------------------------
// Router: pooled mean -> 128 GELU -> 8 logits -> argmax (softmax monotone)
// ---------------------------------------------------------------------------
__global__ void router_kernel(const float* __restrict__ x,
                              const float* __restrict__ w1,
                              const float* __restrict__ b1,
                              const float* __restrict__ w2,
                              const float* __restrict__ b2,
                              int* __restrict__ idx_s, int* __restrict__ idx_t) {
  const int b = blockIdx.x;
  const int t = threadIdx.x;
  __shared__ float pooled[512];
  __shared__ float hbuf[128];
  __shared__ float lg[8];
  for (int c = t * 4; c < t * 4 + 4; ++c) {
    float s = 0.f;
    for (int tt = 0; tt < 64; ++tt) s += x[((size_t)b * 64 + tt) * 512 + c];
    pooled[c] = s * (1.0f / 64.0f);
  }
  __syncthreads();
  {
    float d = b1[t];
    const float* wr = w1 + (size_t)t * 512;
    for (int j = 0; j < 512; ++j) d += pooled[j] * wr[j];
    hbuf[t] = gelu_f(d);
  }
  __syncthreads();
  if (t < 8) {
    float s = b2[t];
    const float* wr = w2 + (size_t)t * 128;
    for (int j = 0; j < 128; ++j) s += hbuf[j] * wr[j];
    lg[t] = s;
  }
  __syncthreads();
  if (t == 0) {
    int bs = 0, bt = 0;
    for (int i = 1; i < 4; ++i) if (lg[i] > lg[bs]) bs = i;
    for (int i = 1; i < 4; ++i) if (lg[4 + i] > lg[4 + bt]) bt = i;
    idx_s[b] = bs;
    idx_t[b] = bt;
  }
}

// ---------------------------------------------------------------------------
// LayerNorm over D=512, one wave per row, 8 rows/block; up to two norms.
// ---------------------------------------------------------------------------
__global__ void ln_kernel(const float* __restrict__ x, int nrows,
                          const float* __restrict__ w1, const float* __restrict__ b1,
                          __bf16* __restrict__ o1,
                          const float* __restrict__ w2, const float* __restrict__ b2,
                          __bf16* __restrict__ o2, float* __restrict__ o2f) {
  const int wave = threadIdx.x >> 5;
  const int lane = threadIdx.x & 31;
  const int row  = blockIdx.x * 8 + wave;
  if (row >= nrows) return;
  const float* xr = x + (size_t)row * 512;
  float v[16];
  float s = 0.f;
#pragma unroll
  for (int i = 0; i < 16; ++i) { v[i] = xr[i * 32 + lane]; s += v[i]; }
#pragma unroll
  for (int m = 16; m >= 1; m >>= 1) s += __shfl_xor(s, m, 32);
  const float mean = s * (1.0f / 512.0f);
  float ss = 0.f;
#pragma unroll
  for (int i = 0; i < 16; ++i) { float d = v[i] - mean; ss += d * d; }
#pragma unroll
  for (int m = 16; m >= 1; m >>= 1) ss += __shfl_xor(ss, m, 32);
  const float inv = rsqrtf(ss * (1.0f / 512.0f) + 1e-5f);
#pragma unroll
  for (int i = 0; i < 16; ++i) {
    const int c = i * 32 + lane;
    const float nv = (v[i] - mean) * inv;
    if (o1) o1[(size_t)row * 512 + c] = f2bf(nv * w1[c] + b1[c]);
    if (o2) {
      const float y2 = nv * w2[c] + b2[c];
      o2[(size_t)row * 512 + c] = f2bf(y2);
      if (o2f) o2f[(size_t)row * 512 + c] = y2;
    }
  }
}

// ---------------------------------------------------------------------------
// Generic 64x64-tile bf16 WMMA GEMM:  C[b] = act(A[b] @ W(idx[b])^T + bias)
// Double-buffered LDS; stage k+1 DMA'd via async-to-LDS while stage k does
// 8 WMMAs.  Block = 128 threads = 4 waves.
// ---------------------------------------------------------------------------
__global__ void gemm_wmma_kernel(const __bf16* __restrict__ A,
                                 const __bf16* __restrict__ W,
                                 const float* __restrict__ bias,
                                 const int* __restrict__ eidx,
                                 int Ntot, int Kd,
                                 const float* __restrict__ res,
                                 float* __restrict__ outf,
                                 __bf16* __restrict__ outb,
                                 int act_gelu) {
  const int b      = blockIdx.z;
  const int tile_n = blockIdx.x * 64;
  const int tid    = threadIdx.x;
  const int lane   = tid & 31;
  const int wave   = tid >> 5;

  __shared__ __align__(16) __bf16 sA[2][64 * 80];  // 64 rows x 64 K, stride 80
  __shared__ __align__(16) __bf16 sB[2][64 * 80];

  const size_t wsel  = eidx ? (size_t)eidx[b] : 0;
  const __bf16* Ab   = A + (size_t)b * 64 * Kd;
  const __bf16* Wb   = W + wsel * (size_t)Ntot * Kd + (size_t)tile_n * Kd;
  const float*  bb   = bias + wsel * (size_t)Ntot;

  const int lr  = tid >> 1;           // row owned by this thread
  const int lkk = (tid & 1) * 32;     // 32-wide K half

  v8f acc[4] = {};
  const int nk = Kd >> 6;             // K chunks of 64

  // issue one stage's async copies (8 x b128 per thread => ASYNCcnt += 8/wave)
  auto issue = [&](int kc, int buf) {
    const __bf16* ga = Ab + (size_t)lr * Kd + kc * 64 + lkk;
    const __bf16* gw = Wb + (size_t)lr * Kd + kc * 64 + lkk;
#pragma unroll
    for (int i = 0; i < 4; ++i) {
      async_ld_b128(&sA[buf][lr * 80 + lkk + i * 8], ga + i * 8);
      async_ld_b128(&sB[buf][lr * 80 + lkk + i * 8], gw + i * 8);
    }
  };

  issue(0, 0);
  for (int kc = 0; kc < nk; ++kc) {
    const int cur = kc & 1;
    if (kc + 1 < nk) {
      issue(kc + 1, cur ^ 1);
      if (kc + 2 < nk)  // far L2 prefetch of weight stream
        __builtin_prefetch(Wb + (size_t)lr * Kd + (kc + 2) * 64 + lkk, 0, 1);
      WAIT_ASYNC(8);    // stage kc complete (in-order ASYNCcnt)
    } else {
      WAIT_ASYNC(0);
    }
    __syncthreads();
#pragma unroll
    for (int kt = 0; kt < 2; ++kt) {
      const v16bf af = load_frag_a(&sA[cur][kt * 32], 80, wave * 16, lane);
#pragma unroll
      for (int nt = 0; nt < 4; ++nt) {
        const v16bf bf = load_frag_b(&sB[cur][kt * 32], 80, nt * 16, lane);
        acc[nt] = __builtin_amdgcn_wmma_f32_16x16x32_bf16(
            false, af, false, bf, (short)0, acc[nt], false, false);
      }
    }
    __syncthreads();  // buffer free before it is overwritten two stages later
  }

  // epilogue: bias (+gelu) (+residual) -> fp32/bf16 stores
  const int mbase = wave * 16 + ((lane & 16) ? 8 : 0);
#pragma unroll
  for (int nt = 0; nt < 4; ++nt) {
    const int n = tile_n + nt * 16 + (lane & 15);
    const float bv = bb[n];
#pragma unroll
    for (int r = 0; r < 8; ++r) {
      const int m = mbase + r;
      float v = acc[nt][r] + bv;
      if (act_gelu) v = gelu_f(v);
      const size_t off = ((size_t)b * 64 + m) * Ntot + n;
      if (res) v += res[off];
      if (outf) outf[off] = v;
      if (outb) outb[off] = f2bf(v);
    }
  }
}

// ---------------------------------------------------------------------------
// Fused single-head attention: block per (head, batch); T=S=HD=64.
// ---------------------------------------------------------------------------
__global__ void attn_kernel(const __bf16* __restrict__ q, int qrs, int qco,
                            const __bf16* __restrict__ k, int krs, int kco,
                            const __bf16* __restrict__ v, int vrs, int vco,
                            __bf16* __restrict__ o, int ors,
                            float scale, int causal) {
  const int h = blockIdx.x;
  const int b = blockIdx.y;
  const int tid  = threadIdx.x;
  const int lane = tid & 31;
  const int wave = tid >> 5;

  __shared__ __align__(16) __bf16 sQ[64 * 72];   // q, later att (bf16)
  __shared__ __align__(16) __bf16 sK[64 * 72];   // [S][HD] == B^T
  __shared__ __align__(16) __bf16 sVT[64 * 72];  // v transposed -> [HD][S]
  __shared__ float sS[64 * 72];

  const size_t rowb = (size_t)b * 64;
  {
    const int r  = tid >> 1;
    const int kk = (tid & 1) * 32;
    const __bf16* gq = q + (rowb + r) * qrs + qco + h * 64 + kk;
    const __bf16* gk = k + (rowb + r) * krs + kco + h * 64 + kk;
#pragma unroll
    for (int i = 0; i < 4; ++i) {
      async_ld_b128(&sQ[r * 72 + kk + i * 8], gq + i * 8);
      async_ld_b128(&sK[r * 72 + kk + i * 8], gk + i * 8);
    }
    const __bf16* gv = v + (rowb + r) * vrs + vco + h * 64 + kk;
    for (int j = 0; j < 32; ++j) sVT[(kk + j) * 72 + r] = gv[j];
    WAIT_ASYNC(0);
  }
  __syncthreads();

  // scores = q @ k^T
  {
    v8f acc[4] = {};
#pragma unroll
    for (int kt = 0; kt < 2; ++kt) {
      const v16bf af = load_frag_a(sQ + kt * 32, 72, wave * 16, lane);
#pragma unroll
      for (int nt = 0; nt < 4; ++nt) {
        const v16bf bf = load_frag_b(sK + kt * 32, 72, nt * 16, lane);
        acc[nt] = __builtin_amdgcn_wmma_f32_16x16x32_bf16(
            false, af, false, bf, (short)0, acc[nt], false, false);
      }
    }
    const int mbase = wave * 16 + ((lane & 16) ? 8 : 0);
#pragma unroll
    for (int nt = 0; nt < 4; ++nt) {
      const int n = nt * 16 + (lane & 15);
#pragma unroll
      for (int r = 0; r < 8; ++r) {
        const int m = mbase + r;
        float sv = acc[nt][r] * scale;
        if (causal && n > m) sv = -3.0e38f;
        sS[m * 72 + n] = sv;
      }
    }
  }
  __syncthreads();

  // softmax rows -> bf16 att stored back into sQ
  if (tid < 64) {
    float mx = -3.0e38f;
    for (int j = 0; j < 64; ++j) mx = fmaxf(mx, sS[tid * 72 + j]);
    float sm = 0.f;
    for (int j = 0; j < 64; ++j) {
      const float e = __expf(sS[tid * 72 + j] - mx);
      sS[tid * 72 + j] = e;
      sm += e;
    }
    const float inv = 1.0f / sm;
    for (int j = 0; j < 64; ++j) sQ[tid * 72 + j] = f2bf(sS[tid * 72 + j] * inv);
  }
  __syncthreads();

  // out = att @ v
  {
    v8f acc[4] = {};
#pragma unroll
    for (int kt = 0; kt < 2; ++kt) {
      const v16bf af = load_frag_a(sQ + kt * 32, 72, wave * 16, lane);
#pragma unroll
      for (int nt = 0; nt < 4; ++nt) {
        const v16bf bf = load_frag_b(sVT + kt * 32, 72, nt * 16, lane);
        acc[nt] = __builtin_amdgcn_wmma_f32_16x16x32_bf16(
            false, af, false, bf, (short)0, acc[nt], false, false);
      }
    }
    const int mbase = wave * 16 + ((lane & 16) ? 8 : 0);
#pragma unroll
    for (int nt = 0; nt < 4; ++nt) {
      const int n = nt * 16 + (lane & 15);
#pragma unroll
      for (int r = 0; r < 8; ++r) {
        const int m = mbase + r;
        o[(rowb + m) * ors + h * 64 + n] = f2bf(acc[nt][r]);
      }
    }
  }
}

// ---------------------------------------------------------------------------
extern "C" void kernel_launch(void* const* d_in, const int* in_sizes, int n_in,
                              void* d_out, int out_size, void* d_ws, size_t ws_size,
                              hipStream_t stream) {
  (void)in_sizes; (void)n_in; (void)out_size; (void)ws_size;
  const int B = 256, Hh = 8;
  const int rows = B * 64;  // 16384

  const float* x        = (const float*)d_in[0];
  const float* rw1      = (const float*)d_in[1];
  const float* rb1      = (const float*)d_in[2];
  const float* rw2      = (const float*)d_in[3];
  const float* rb2      = (const float*)d_in[4];
  const float* nsw      = (const float*)d_in[5];
  const float* nsb      = (const float*)d_in[6];
  const float* ntw      = (const float*)d_in[7];
  const float* ntb      = (const float*)d_in[8];
  const float* sp_in_w  = (const float*)d_in[9];
  const float* sp_in_b  = (const float*)d_in[10];
  const float* sp_out_w = (const float*)d_in[11];
  const float* sp_out_b = (const float*)d_in[12];
  const float* tq_w     = (const float*)d_in[13];
  const float* tq_b     = (const float*)d_in[14];
  const float* tk_w     = (const float*)d_in[15];
  const float* tk_b     = (const float*)d_in[16];
  const float* tv_w     = (const float*)d_in[17];
  const float* tv_b     = (const float*)d_in[18];
  const float* tp_w     = (const float*)d_in[19];
  const float* tp_b     = (const float*)d_in[20];
  const float* cin_w    = (const float*)d_in[21];
  const float* cin_b    = (const float*)d_in[22];
  const float* cout_w   = (const float*)d_in[23];
  const float* cout_b   = (const float*)d_in[24];
  const float* nmw      = (const float*)d_in[25];
  const float* nmb      = (const float*)d_in[26];
  const float* mw1      = (const float*)d_in[27];
  const float* mb1      = (const float*)d_in[28];
  const float* mw2      = (const float*)d_in[29];
  const float* mb2      = (const float*)d_in[30];

  // ---- workspace layout: 4KB idx | 160MB activations | ~24MB bf16 weights --
  char* ws = (char*)d_ws;
  int* idx_s = (int*)ws;
  int* idx_t = idx_s + 256;
  const size_t MB = 1024 * 1024;
  char* base = ws + 4096;
  __bf16* bufA = (__bf16*)(base);                 // 16 MB
  __bf16* bufB = (__bf16*)(base + 16 * MB);       // 16 MB
  float*  bufC = (float*) (base + 32 * MB);       // 32 MB
  __bf16* bufD = (__bf16*)(base + 64 * MB);       // 48 MB (+E gives 96)
  __bf16* bufE = (__bf16*)(base + 112 * MB);      // 48 MB
  __bf16* wbase = (__bf16*)(base + 160 * MB);     // bf16 weights

  __bf16* W_spin  = wbase;                         // 4*1536*512
  __bf16* W_spout = W_spin  + 3145728;             // 4*512*512
  __bf16* W_tq    = W_spout + 1048576;
  __bf16* W_tk    = W_tq    + 1048576;
  __bf16* W_tv    = W_tk    + 1048576;
  __bf16* W_tp    = W_tv    + 1048576;
  __bf16* W_cin   = W_tp    + 1048576;             // 1536*512
  __bf16* W_cout  = W_cin   + 786432;              // 512*512
  __bf16* W_m1    = W_cout  + 262144;              // 2048*512
  __bf16* W_m2    = W_m1    + 1048576;             // 512*2048

  __bf16* xn  = bufA;
  __bf16* xt  = bufB;
  float*  xtf = bufC;
  __bf16* qkv = bufD;
  __bf16* osp = bufA;
  __bf16* spo = bufD;
  __bf16* tq  = bufE;
  __bf16* tk  = bufE + (size_t)8 * MB;
  __bf16* tv  = bufE + (size_t)16 * MB;
  __bf16* ot  = bufA;
  __bf16* tmo = bufB;
  __bf16* cq  = bufE;
  __bf16* ck  = bufE + (size_t)8 * MB;
  __bf16* cv  = bufE + (size_t)16 * MB;
  __bf16* oc  = bufA;
  float*  x1  = bufC;
  __bf16* h2  = bufA;
  __bf16* hm  = bufD;  // [B,T,2048] (64MB: D+E)

  const float scale = 0.125f;  // HD^-0.5
  const dim3 blk128(128), blk256(256);
  auto cvt = [&](const float* s, __bf16* d, int n) {
    cvt_bf16_kernel<<<(n / 8 + 255) / 256, blk256, 0, stream>>>(s, d, n);
  };

  // 0. weight conversion passes
  cvt(sp_in_w,  W_spin,  3145728);
  cvt(sp_out_w, W_spout, 1048576);
  cvt(tq_w,     W_tq,    1048576);
  cvt(tk_w,     W_tk,    1048576);
  cvt(tv_w,     W_tv,    1048576);
  cvt(tp_w,     W_tp,    1048576);
  cvt(cin_w,    W_cin,   786432);
  cvt(cout_w,   W_cout,  262144);
  cvt(mw1,      W_m1,    1048576);
  cvt(mw2,      W_m2,    1048576);

  // 1. router
  router_kernel<<<B, blk128, 0, stream>>>(x, rw1, rb1, rw2, rb2, idx_s, idx_t);
  // 2. dual LN
  ln_kernel<<<rows / 8, blk256, 0, stream>>>(x, rows, nsw, nsb, xn, ntw, ntb, xt, xtf);
  // 3. spatial qkv (expert idx_s)
  gemm_wmma_kernel<<<dim3(24, 1, B), blk128, 0, stream>>>(
      xn, W_spin, sp_in_b, idx_s, 1536, 512, nullptr, nullptr, qkv, 0);
  // 4. spatial attention
  attn_kernel<<<dim3(Hh, B), blk128, 0, stream>>>(
      qkv, 1536, 0, qkv, 1536, 512, qkv, 1536, 1024, osp, 512, scale, 0);
  // 5. spatial out proj
  gemm_wmma_kernel<<<dim3(8, 1, B), blk128, 0, stream>>>(
      osp, W_spout, sp_out_b, idx_s, 512, 512, nullptr, nullptr, spo, 0);
  // 6. temporal q/k/v projections (expert idx_t)
  gemm_wmma_kernel<<<dim3(8, 1, B), blk128, 0, stream>>>(
      xt, W_tq, tq_b, idx_t, 512, 512, nullptr, nullptr, tq, 0);
  gemm_wmma_kernel<<<dim3(8, 1, B), blk128, 0, stream>>>(
      xt, W_tk, tk_b, idx_t, 512, 512, nullptr, nullptr, tk, 0);
  gemm_wmma_kernel<<<dim3(8, 1, B), blk128, 0, stream>>>(
      xt, W_tv, tv_b, idx_t, 512, 512, nullptr, nullptr, tv, 0);
  // 7. temporal causal attention
  attn_kernel<<<dim3(Hh, B), blk128, 0, stream>>>(
      tq, 512, 0, tk, 512, 0, tv, 512, 0, ot, 512, scale, 1);
  // 8. temporal out proj + residual xt
  gemm_wmma_kernel<<<dim3(8, 1, B), blk128, 0, stream>>>(
      ot, W_tp, tp_b, idx_t, 512, 512, xtf, nullptr, tmo, 0);
  // 9. cross q/k/v
  gemm_wmma_kernel<<<dim3(8, 1, B), blk128, 0, stream>>>(
      spo, W_cin, cin_b, nullptr, 512, 512, nullptr, nullptr, cq, 0);
  gemm_wmma_kernel<<<dim3(8, 1, B), blk128, 0, stream>>>(
      tmo, W_cin + (size_t)512 * 512, cin_b + 512, nullptr, 512, 512,
      nullptr, nullptr, ck, 0);
  gemm_wmma_kernel<<<dim3(8, 1, B), blk128, 0, stream>>>(
      tmo, W_cin + (size_t)1024 * 512, cin_b + 1024, nullptr, 512, 512,
      nullptr, nullptr, cv, 0);
  // 10. cross attention
  attn_kernel<<<dim3(Hh, B), blk128, 0, stream>>>(
      cq, 512, 0, ck, 512, 0, cv, 512, 0, oc, 512, scale, 0);
  // 11. cross out proj + residual x -> x1
  gemm_wmma_kernel<<<dim3(8, 1, B), blk128, 0, stream>>>(
      oc, W_cout, cout_b, nullptr, 512, 512, x, x1, nullptr, 0);
  // 12. MLP layernorm
  ln_kernel<<<rows / 8, blk256, 0, stream>>>(x1, rows, nmw, nmb, h2,
                                             nullptr, nullptr, nullptr, nullptr);
  // 13. MLP up (gelu)
  gemm_wmma_kernel<<<dim3(32, 1, B), blk128, 0, stream>>>(
      h2, W_m1, mb1, nullptr, 2048, 512, nullptr, nullptr, hm, 1);
  // 14. MLP down + residual x1 -> d_out
  gemm_wmma_kernel<<<dim3(8, 1, B), blk128, 0, stream>>>(
      hm, W_m2, mb2, nullptr, 512, 2048, x1, (float*)d_out, nullptr, 0);
}
